// NonlinearMixtureMobile_30983894073533
// MI455X (gfx1250) — matrix operand order repaired and checked by reading, not compiled, not run on hardware
//
#include <hip/hip_runtime.h>

// ---------------------------------------------------------------------------
// CDNA5 (gfx1250) MoE-MobileNetV2, top-1 gating collapsed to one expert/image.
// Activations: NHWC bf16, channels padded to 32 with hard zeros.
// 1x1 convs: v_wmma_f32_16x16x32_bf16, weights pre-swizzled to fragment order.
// 2x2 register tiling (32ch x 32px per wave) -> 2 b128 loads per WMMA.
// ---------------------------------------------------------------------------

typedef __attribute__((ext_vector_type(16))) __bf16          v16bf;
typedef __attribute__((ext_vector_type(16))) unsigned short  v16u;
typedef __attribute__((ext_vector_type(8)))  unsigned int    v8u;
typedef __attribute__((ext_vector_type(8)))  float           v8f;

struct P8 { const float* p[8]; };   // one pointer per expert (kernarg, by value)

#define BN_SCALE 0.99999500003750f  // 1/sqrt(1+1e-5), eval-mode BN fold

__device__ __forceinline__ unsigned short f2bf(float f) {
  unsigned u = __float_as_uint(f);
  u += 0x7FFFu + ((u >> 16) & 1u);          // round-to-nearest-even
  return (unsigned short)(u >> 16);
}
__device__ __forceinline__ float bf2f(unsigned short h) {
  return __uint_as_float(((unsigned)h) << 16);
}

// ---------------------------------------------------------------------------
// Router: logits[b][e] = sum_{c,i,j} x[b,c,i,j] * w[e,c,i%4,j%4] + 64*bias[e]
// ---------------------------------------------------------------------------
__global__ __launch_bounds__(256) void moe_router(
    const float* __restrict__ x, const float* __restrict__ rw,
    const float* __restrict__ rb, float* __restrict__ selB,
    float* __restrict__ gateB, int* __restrict__ idxB,
    float* __restrict__ dispatchOut)
{
  __shared__ float red[256];
  __shared__ float logits[8];
  const int b = blockIdx.x, t = threadIdx.x;
  float part[8];
#pragma unroll
  for (int e = 0; e < 8; ++e) part[e] = 0.f;
  for (int n = t; n < 3072; n += 256) {
    const int c = n >> 10, r = n & 1023, i = (r >> 5) & 3, j = r & 3;
    const float xv = x[b * 3072 + n];
#pragma unroll
    for (int e = 0; e < 8; ++e)
      part[e] += xv * rw[((e * 3 + c) * 4 + i) * 4 + j];
  }
  for (int e = 0; e < 8; ++e) {
    red[t] = part[e]; __syncthreads();
    for (int s = 128; s > 0; s >>= 1) {
      if (t < s) red[t] += red[t + s];
      __syncthreads();
    }
    if (t == 0) logits[e] = red[0] + 64.f * rb[e];
    __syncthreads();
  }
  if (t == 0) {
    float mx = -1e30f; int am = 0;
    for (int e = 0; e < 8; ++e) if (logits[e] > mx) { mx = logits[e]; am = e; }
    float ex[8], s = 0.f;
    for (int e = 0; e < 8; ++e) { ex[e] = __expf(logits[e] - mx); s += ex[e]; }
    const float inv = 1.f / s;
    for (int e = 0; e < 8; ++e) {
      selB[b * 8 + e] = ex[e] * inv;
      dispatchOut[b * 8 + e] = (e == am) ? 1.f : 0.f;
    }
    gateB[b] = ex[am] * inv;
    idxB[b]  = am;
  }
}

// loss = mean_e(density_proxy*density)*E^2 = sum_e(dp_e*d_e)*8
__global__ __launch_bounds__(64) void moe_loss(
    const float* __restrict__ selB, const int* __restrict__ idxB,
    float* __restrict__ outTail)
{
  __shared__ float prod[8];
  const int t = threadIdx.x;
  if (t < 8) {
    float dp = 0.f, d = 0.f;
    for (int b = 0; b < 64; ++b) {
      dp += selB[b * 8 + t];
      d  += (idxB[b] == t) ? 1.f : 0.f;
    }
    prod[t] = (dp * (1.f / 64.f)) * (d * (1.f / 64.f));
  }
  __syncthreads();
  if (t == 0) {
    float s = 0.f;
    for (int e = 0; e < 8; ++e) s += prod[e];
    outTail[0] = s * 8.f;
    outTail[1] = 0.f;
  }
}

// ---------------------------------------------------------------------------
// Weight pre-swizzle: fp32 [Cout,Cin] -> bf16 A-fragment tiles.
// Tile (mt,kt) holds 16x32; lane l keeps 16 contiguous bf16:
//   element i <-> k = kt*32 + 16*(l>>4) + i, m = mt*16 + (l&15)
// ---------------------------------------------------------------------------
__global__ __launch_bounds__(32) void moe_prep(
    P8 W, unsigned short* __restrict__ dst, int Cout, int Cin, int Kt,
    long layerElems)
{
  const int e = blockIdx.y;
  const float* __restrict__ w = W.p[e];
  const int tile = blockIdx.x;               // mt*Kt + kt
  const int mt = tile / Kt, kt = tile % Kt;
  const int l = threadIdx.x, half = l >> 4, lm = l & 15;
  const int m  = mt * 16 + lm;
  const int mc = m < Cout ? m : Cout - 1;    // clamped (always in-bounds)
  v16u vals;
#pragma unroll
  for (int i = 0; i < 16; ++i) {
    const int k  = kt * 32 + 16 * half + i;
    const int kc = k < Cin ? k : Cin - 1;
    float v = w[(size_t)mc * Cin + kc];      // unconditional load
    v = (m < Cout && k < Cin) ? v : 0.f;     // v_cndmask, no branch
    vals[i] = f2bf(v);
  }
  *(v8u*)(dst + (size_t)e * layerElems + (size_t)tile * 512 + l * 16) =
      __builtin_bit_cast(v8u, vals);
}

// ---------------------------------------------------------------------------
// Shared epilogue: BN (+residual) (+ReLU), channel padding stored as hard 0,
// adjacent-channel pairs packed into single b32 stores.
// ---------------------------------------------------------------------------
__device__ __forceinline__ void pw_store(
    const v8f acc, const float* __restrict__ g, const float* __restrict__ bt,
    unsigned short* __restrict__ op, const unsigned short* __restrict__ rp,
    int mbase, int Cout, int relu)
{
#pragma unroll
  for (int r = 0; r < 8; r += 2) {
    float v0, v1;
    {
      const int mo = mbase + r;
      const int mc = mo < Cout ? mo : Cout - 1;
      float v = acc[r] * (g[mc] * BN_SCALE) + bt[mc];
      if (rp) v += bf2f(rp[r]);
      if (relu) v = fmaxf(v, 0.f);
      v0 = (mo < Cout) ? v : 0.f;
    }
    {
      const int mo = mbase + r + 1;
      const int mc = mo < Cout ? mo : Cout - 1;
      float v = acc[r + 1] * (g[mc] * BN_SCALE) + bt[mc];
      if (rp) v += bf2f(rp[r + 1]);
      if (relu) v = fmaxf(v, 0.f);
      v1 = (mo < Cout) ? v : 0.f;
    }
    *(unsigned*)(op + r) = (unsigned)f2bf(v0) | ((unsigned)f2bf(v1) << 16);
  }
}

// ---------------------------------------------------------------------------
// Pointwise conv GEMM, 16x16 tile / wave (used only for HW==16 layers).
// ---------------------------------------------------------------------------
__global__ __launch_bounds__(32) void moe_pw(
    const unsigned short* __restrict__ wprep, long layerElems,
    P8 G, P8 Bv,
    const unsigned short* __restrict__ actIn, unsigned short* __restrict__ actOut,
    const unsigned short* __restrict__ resid, const int* __restrict__ idx,
    int Cinp, int Cout, int Coutp, int HW, int Kt, int relu,
    long inStride, long outStride, long resStride)
{
  const int b = blockIdx.z;
  const int e = idx[b];
  const int l = threadIdx.x, half = l >> 4, lm = l & 15;
  const int mt = blockIdx.x;
  const int p  = blockIdx.y * 16 + lm;

  const unsigned short* __restrict__ wp =
      wprep + (size_t)e * layerElems + (size_t)mt * Kt * 512 + l * 16;
  const unsigned short* __restrict__ bp =
      actIn + (size_t)b * inStride + (size_t)p * Cinp + 16 * half;

  v8f acc = {};
  for (int kt = 0; kt < Kt; ++kt) {
    __builtin_prefetch(wp + 512, 0, 1);                       // global_prefetch_b8
    const v16bf A = __builtin_bit_cast(v16bf, *(const v8u*)wp);
    const v16bf B = __builtin_bit_cast(v16bf, *(const v8u*)bp);
    acc = __builtin_amdgcn_wmma_f32_16x16x32_bf16(
        false, A, false, B, (short)0, acc, false, false);
    wp += 512;
    bp += 32;
  }

  const int mbase = mt * 16 + 8 * half;
  unsigned short* __restrict__ op =
      actOut + (size_t)b * outStride + (size_t)p * Coutp + mbase;
  const unsigned short* __restrict__ rp =
      resid ? resid + (size_t)b * resStride + (size_t)p * Coutp + mbase : nullptr;
  pw_store(acc, G.p[e], Bv.p[e], op, rp, mbase, Cout, relu);
}

// ---------------------------------------------------------------------------
// Pointwise conv GEMM, 2x2 tiles (32ch x 32px) / wave: 8 b128 loads feed
// 4 independent WMMAs -> 2 loads/WMMA and no accumulator RAW stalls.
// ---------------------------------------------------------------------------
__global__ __launch_bounds__(32) void moe_pw2(
    const unsigned short* __restrict__ wprep, long layerElems,
    P8 G, P8 Bv,
    const unsigned short* __restrict__ actIn, unsigned short* __restrict__ actOut,
    const unsigned short* __restrict__ resid, const int* __restrict__ idx,
    int Cinp, int Cout, int Coutp, int HW, int Kt, int relu,
    long inStride, long outStride, long resStride)
{
  const int b = blockIdx.z;
  const int e = idx[b];
  const int l = threadIdx.x, half = l >> 4, lm = l & 15;
  const int mt0 = blockIdx.x * 2;
  const int p0  = blockIdx.y * 32;
  const int p   = p0 + lm;

  const unsigned short* __restrict__ wp0 =
      wprep + (size_t)e * layerElems + (size_t)mt0 * Kt * 512 + l * 16;
  const unsigned short* __restrict__ wp1 = wp0 + (size_t)Kt * 512;
  const unsigned short* __restrict__ bp0 =
      actIn + (size_t)b * inStride + (size_t)p * Cinp + 16 * half;
  const unsigned short* __restrict__ bp1 = bp0 + (size_t)16 * Cinp;

  v8f a00 = {}, a01 = {}, a10 = {}, a11 = {};
  for (int kt = 0; kt < Kt; ++kt) {
    __builtin_prefetch(wp0 + 512, 0, 1);
    __builtin_prefetch(wp1 + 512, 0, 1);
    const v16bf A0 = __builtin_bit_cast(v16bf, *(const v8u*)wp0);
    const v16bf A1 = __builtin_bit_cast(v16bf, *(const v8u*)wp1);
    const v16bf B0 = __builtin_bit_cast(v16bf, *(const v8u*)bp0);
    const v16bf B1 = __builtin_bit_cast(v16bf, *(const v8u*)bp1);
    a00 = __builtin_amdgcn_wmma_f32_16x16x32_bf16(false, A0, false, B0, (short)0, a00, false, false);
    a01 = __builtin_amdgcn_wmma_f32_16x16x32_bf16(false, A0, false, B1, (short)0, a01, false, false);
    a10 = __builtin_amdgcn_wmma_f32_16x16x32_bf16(false, A1, false, B0, (short)0, a10, false, false);
    a11 = __builtin_amdgcn_wmma_f32_16x16x32_bf16(false, A1, false, B1, (short)0, a11, false, false);
    wp0 += 512; wp1 += 512; bp0 += 32; bp1 += 32;
  }

  const float* __restrict__ g  = G.p[e];
  const float* __restrict__ bt = Bv.p[e];
  unsigned short* __restrict__ ob = actOut + (size_t)b * outStride;
  const unsigned short* __restrict__ rbp =
      resid ? resid + (size_t)b * resStride : nullptr;
#pragma unroll
  for (int ti = 0; ti < 2; ++ti) {
    const int mbase = (mt0 + ti) * 16 + 8 * half;
#pragma unroll
    for (int tj = 0; tj < 2; ++tj) {
      const int pp = p0 + tj * 16 + lm;
      unsigned short* op = ob + (size_t)pp * Coutp + mbase;
      const unsigned short* rp = rbp ? rbp + (size_t)pp * Coutp + mbase : nullptr;
      const v8f acc = (ti == 0) ? (tj == 0 ? a00 : a01) : (tj == 0 ? a10 : a11);
      pw_store(acc, g, bt, op, rp, mbase, Cout, relu);
    }
  }
}

// ---------------------------------------------------------------------------
// Stem conv 3x3 s1 p1, cin=3, cout=32, fp32 NCHW in -> bf16 NHWC out.
// ---------------------------------------------------------------------------
__global__ __launch_bounds__(256) void moe_conv1(
    P8 W, P8 G, P8 Bv, const float* __restrict__ x,
    unsigned short* __restrict__ out, const int* __restrict__ idx, long outStride)
{
  const int b = blockIdx.z;
  const int e = idx[b];
  const int t = blockIdx.x * 256 + threadIdx.x;
  if (t >= 32768) return;                    // 1024 pixels * 32 channels
  const int c = t & 31, pix = t >> 5, y = pix >> 5, xx = pix & 31;
  const float* __restrict__ w  = W.p[e] + c * 27;
  const float* __restrict__ xi = x + b * 3072;
  float acc = 0.f;
#pragma unroll
  for (int ci = 0; ci < 3; ++ci)
#pragma unroll
    for (int dy = 0; dy < 3; ++dy) {
      const int iy  = y + dy - 1;
      const int iyc = iy < 0 ? 0 : (iy > 31 ? 31 : iy);
      const bool yok = (unsigned)iy < 32u;
#pragma unroll
      for (int dx = 0; dx < 3; ++dx) {
        const int ix  = xx + dx - 1;
        const int ixc = ix < 0 ? 0 : (ix > 31 ? 31 : ix);
        const float wv = (yok && (unsigned)ix < 32u) ? w[ci * 9 + dy * 3 + dx] : 0.f;
        acc += xi[ci * 1024 + iyc * 32 + ixc] * wv;
      }
    }
  float v = acc * (G.p[e][c] * BN_SCALE) + Bv.p[e][c];
  v = fmaxf(v, 0.f);
  out[(size_t)b * outStride + t] = f2bf(v);
}

// ---------------------------------------------------------------------------
// Depthwise 3x3 (pad 1, stride s) + BN + ReLU, bf16 NHWC, padded channels -> 0.
// ---------------------------------------------------------------------------
__global__ __launch_bounds__(256) void moe_dw(
    P8 W, P8 G, P8 Bv,
    const unsigned short* __restrict__ in_, unsigned short* __restrict__ out,
    const int* __restrict__ idx, int C, int Cp, int Hin, int stride,
    long inStride, long outStride)
{
  const int b = blockIdx.z;
  const int e = idx[b];
  const int Hout = Hin / stride;
  const int t = blockIdx.x * 256 + threadIdx.x;
  if (t >= Cp * Hout * Hout) return;
  const int c = t % Cp, pix = t / Cp, oy = pix / Hout, ox = pix % Hout;
  const int cc = c < C ? c : C - 1;
  const float* __restrict__ w = W.p[e] + cc * 9;
  const unsigned short* __restrict__ ip = in_ + (size_t)b * inStride;
  float acc = 0.f;
#pragma unroll
  for (int dy = 0; dy < 3; ++dy) {
    const int iy  = oy * stride + dy - 1;
    const int iyc = iy < 0 ? 0 : (iy >= Hin ? Hin - 1 : iy);
    const bool yok = (unsigned)iy < (unsigned)Hin;
#pragma unroll
    for (int dx = 0; dx < 3; ++dx) {
      const int ix  = ox * stride + dx - 1;
      const int ixc = ix < 0 ? 0 : (ix >= Hin ? Hin - 1 : ix);
      const float wv = (yok && (unsigned)ix < (unsigned)Hin) ? w[dy * 3 + dx] : 0.f;
      acc += bf2f(ip[(size_t)(iyc * Hin + ixc) * Cp + c]) * wv;
    }
  }
  float v = acc * (G.p[e][cc] * BN_SCALE) + Bv.p[e][cc];
  v = fmaxf(v, 0.f);
  v = (c < C) ? v : 0.f;                     // keep channel padding exactly 0
  out[(size_t)b * outStride + t] = f2bf(v);
}

// ---------------------------------------------------------------------------
// Head: avgpool(4x4) -> fc(1280->10) -> *gate -> softmax  (NHWC input)
// ---------------------------------------------------------------------------
__global__ __launch_bounds__(256) void moe_head(
    P8 FW, P8 FB, const unsigned short* __restrict__ act,
    const float* __restrict__ gate, const int* __restrict__ idx,
    float* __restrict__ outP, long actStride)
{
  __shared__ float pooled[1280];
  __shared__ float red[256];
  __shared__ float logit[10];
  const int b = blockIdx.x;
  const int e = idx[b];
  const float* __restrict__ fw = FW.p[e];
  const float* __restrict__ fb = FB.p[e];
  const unsigned short* __restrict__ a = act + (size_t)b * actStride;
  for (int c = threadIdx.x; c < 1280; c += 256) {
    float s = 0.f;
#pragma unroll
    for (int q = 0; q < 16; ++q) s += bf2f(a[q * 1280 + c]);
    pooled[c] = s * (1.f / 16.f);
  }
  __syncthreads();
  for (int cls = 0; cls < 10; ++cls) {
    float s = 0.f;
    for (int c = threadIdx.x; c < 1280; c += 256) s += pooled[c] * fw[c * 10 + cls];
    red[threadIdx.x] = s; __syncthreads();
    for (int st = 128; st > 0; st >>= 1) {
      if (threadIdx.x < st) red[threadIdx.x] += red[threadIdx.x + st];
      __syncthreads();
    }
    if (threadIdx.x == 0) logit[cls] = (red[0] + fb[cls]) * gate[b];
    __syncthreads();
  }
  if (threadIdx.x == 0) {
    float mx = -1e30f;
    for (int cls = 0; cls < 10; ++cls) mx = fmaxf(mx, logit[cls]);
    float ex[10], s = 0.f;
    for (int cls = 0; cls < 10; ++cls) { ex[cls] = __expf(logit[cls] - mx); s += ex[cls]; }
    const float inv = 1.f / s;
    for (int cls = 0; cls < 10; ++cls) outP[b * 10 + cls] = ex[cls] * inv;
  }
}

// ---------------------------------------------------------------------------
// Host orchestration
// ---------------------------------------------------------------------------
static inline int ru32(int c) { return (c + 31) & ~31; }

extern "C" void kernel_launch(void* const* d_in, const int* in_sizes, int n_in,
                              void* d_out, int out_size, void* d_ws, size_t ws_size,
                              hipStream_t stream)
{
  (void)in_sizes; (void)n_in; (void)out_size; (void)ws_size;
  const float* x  = (const float*)d_in[0];
  const float* rw = (const float*)d_in[1];
  const float* rb = (const float*)d_in[2];

  struct BM { int s, sc, cin, cout, mid; };
  static const BM META[17] = {
    {1,1, 32, 16,  32},
    {1,1, 16, 24,  96},{1,0, 24, 24, 144},
    {2,0, 24, 32, 144},{1,0, 32, 32, 192},{1,0, 32, 32, 192},
    {2,0, 32, 64, 192},{1,0, 64, 64, 384},{1,0, 64, 64, 384},{1,0, 64, 64, 384},
    {1,1, 64, 96, 384},{1,0, 96, 96, 576},{1,0, 96, 96, 576},
    {2,0, 96,160, 576},{1,0,160,160, 960},{1,0,160,160, 960},
    {1,1,160,320, 960}
  };
  int leafOf[17];
  {
    int off = 3;
    for (int i = 0; i < 17; ++i) { leafOf[i] = off; off += META[i].sc ? 12 : 9; }
    // off == 168: conv2=168, bn2_g=169, bn2_b=170, fc_w=171, fc_b=172
  }
  auto Lp = [&](int e, int leaf) { return (const float*)d_in[3 + e * 173 + leaf]; };
  auto mk = [&](int leaf) { P8 r; for (int e = 0; e < 8; ++e) r.p[e] = Lp(e, leaf); return r; };

  // ---- workspace layout ----
  char*  ws    = (char*)d_ws;
  int*   idxB  = (int*)ws;
  float* gateB = (float*)(ws + 256);
  float* selB  = (float*)(ws + 512);
  const size_t IOE  = 32768;    // io ping-pong (<= 32ch*32*32, NHWC, padded)
  const size_t BIGE = 163840;   // expanded mids (<= 160ch*32*32)
  const size_t SE   = 32768;    // shortcut branch
  unsigned short* bufA = (unsigned short*)(ws + 4096);
  unsigned short* bufB = bufA + (size_t)64 * IOE;
  unsigned short* bufX = bufB + (size_t)64 * IOE;
  unsigned short* bufY = bufX + (size_t)64 * BIGE;
  unsigned short* bufS = bufY + (size_t)64 * BIGE;
  unsigned short* wbase = bufS + (size_t)64 * SE;   // pre-swizzled weights

  float* outProb     = (float*)d_out;      // [64,10]
  float* dispatchOut = outProb + 640;      // [64,8]
  float* lossOut     = outProb + 1152;     // loss, 0.0

  // ---- register + launch all weight preps ----
  struct PW { size_t off; long layerElems; int Kt, Mt; };
  PW pws[40]; int npw = 0; size_t woff = 0;
  auto addPw = [&](int leaf, int Cin, int Cout) -> int {
    const int Kt = ru32(Cin) / 32, Mt = ru32(Cout) / 16;
    const long layerElems = (long)Mt * Kt * 512;
    pws[npw] = { woff, layerElems, Kt, Mt };
    moe_prep<<<dim3(Mt * Kt, 8), 32, 0, stream>>>(
        mk(leaf), wbase + woff, Cout, Cin, Kt, layerElems);
    woff += (size_t)8 * layerElems;
    return npw++;
  };
  int pw1Id[17], pw2Id[17], shId[17];
  for (int i = 0; i < 17; ++i) {
    const BM& m = META[i];
    const int bl = leafOf[i];
    pw1Id[i] = addPw(bl + 0, m.cin, m.mid);
    pw2Id[i] = addPw(bl + 6, m.mid, m.cout);
    shId[i]  = m.sc ? addPw(bl + 9, m.cin, m.cout) : -1;
  }
  const int conv2Id = addPw(168, 320, 1280);

  // unified pw launcher: 2x2-tiled kernel whenever there are >=2 pixel tiles
  auto launchPw = [&](const PW& pw, P8 G, P8 Bv, const unsigned short* in,
                      unsigned short* out, const unsigned short* res,
                      int Cinp, int Cout, int Coutp, int HW, int relu,
                      long inStr, long outStr, long resStr) {
    if (HW >= 32)
      moe_pw2<<<dim3(pw.Mt / 2, HW / 32, 64), 32, 0, stream>>>(
          wbase + pw.off, pw.layerElems, G, Bv, in, out, res, idxB,
          Cinp, Cout, Coutp, HW, pw.Kt, relu, inStr, outStr, resStr);
    else
      moe_pw<<<dim3(pw.Mt, HW / 16, 64), 32, 0, stream>>>(
          wbase + pw.off, pw.layerElems, G, Bv, in, out, res, idxB,
          Cinp, Cout, Coutp, HW, pw.Kt, relu, inStr, outStr, resStr);
  };

  // ---- inference ----
  moe_router<<<64, 256, 0, stream>>>(x, rw, rb, selB, gateB, idxB, dispatchOut);
  moe_loss<<<1, 64, 0, stream>>>(selB, idxB, lossOut);
  moe_conv1<<<dim3(128, 1, 64), 256, 0, stream>>>(mk(0), mk(1), mk(2), x, bufA,
                                                  idxB, (long)IOE);

  unsigned short* cinBuf  = bufA;
  unsigned short* coutBuf = bufB;
  int cur = 32;
  for (int i = 0; i < 17; ++i) {
    const BM& m = META[i];
    const int Hin = cur, Hout = Hin / m.s;
    const int hwin = Hin * Hin, hwout = Hout * Hout;
    const int bl = leafOf[i];
    const int cinp = ru32(m.cin), midp = ru32(m.mid), coutp = ru32(m.cout);
    // expand (pw1) + BN + ReLU
    launchPw(pws[pw1Id[i]], mk(bl + 1), mk(bl + 2), cinBuf, bufX, nullptr,
             cinp, m.mid, midp, hwin, 1, (long)IOE, (long)BIGE, 0L);
    // depthwise 3x3 + BN + ReLU
    {
      const int tot = midp * hwout;
      moe_dw<<<dim3((tot + 255) / 256, 1, 64), 256, 0, stream>>>(
          mk(bl + 3), mk(bl + 4), mk(bl + 5), bufX, bufY, idxB,
          m.mid, midp, Hin, m.s, (long)BIGE, (long)BIGE);
    }
    // residual source
    const unsigned short* resid = nullptr; long rstr = 0;
    if (m.s == 1) {
      if (m.sc) {
        launchPw(pws[shId[i]], mk(bl + 10), mk(bl + 11), cinBuf, bufS, nullptr,
                 cinp, m.cout, coutp, hwin, 0, (long)IOE, (long)SE, 0L);
        resid = bufS; rstr = (long)SE;
      } else {
        resid = cinBuf; rstr = (long)IOE;   // cin==cout here, same padding
      }
    }
    // project (pw2) + BN (+residual), no ReLU
    launchPw(pws[pw2Id[i]], mk(bl + 7), mk(bl + 8), bufY, coutBuf, resid,
             midp, m.cout, coutp, hwout, 0, (long)BIGE, (long)IOE, rstr);
    unsigned short* tmp = cinBuf; cinBuf = coutBuf; coutBuf = tmp;
    cur = Hout;
  }
  // conv2: 320 -> 1280 at 4x4 (HW=16) + BN + ReLU
  launchPw(pws[conv2Id], mk(169), mk(170), cinBuf, bufX, nullptr,
           320, 1280, 1280, 16, 1, (long)IOE, (long)BIGE, 0L);
  // avgpool + fc + gate + softmax
  moe_head<<<64, 256, 0, stream>>>(mk(171), mk(172), bufX, gateB, idxB,
                                   outProb, (long)BIGE);
}
